// RGBD_MultiHeadAttention_46093589021154
// MI455X (gfx1250) — matrix operand order repaired and checked by reading, not compiled, not run on hardware
//
#include <hip/hip_runtime.h>

typedef __bf16 bf16_t;
typedef __attribute__((ext_vector_type(16))) __bf16 v16bf;
typedef __attribute__((ext_vector_type(8)))  __bf16 v8bf;
typedef __attribute__((ext_vector_type(8)))  float  v8f;

#define WMMA_BF16(a, b, c) \
  __builtin_amdgcn_wmma_f32_16x16x32_bf16(false, (a), false, (b), (short)0, (c), false, false)

// Problem constants
constexpr int BATCH = 4, SEQ = 1024, HEADS = 12, HDIM = 64, HID = 768;
constexpr int M_TOT = BATCH * SEQ;     // 4096 rows per modality
constexpr int NQKV  = 3 * HID;         // 2304
constexpr int NQK   = 2 * HID;         // 1536 (q,k columns kept row-major)

// Workspace layout in bf16 elements (all byte offsets multiples of 256)
constexpr size_t OFF_XBF   = 0;                                   // [2][4096][768]
constexpr size_t SZ_XBF    = (size_t)2 * M_TOT * HID;
constexpr size_t OFF_QKVWT = OFF_XBF + SZ_XBF;                    // [2304][768]
constexpr size_t SZ_QKVWT  = (size_t)NQKV * HID;
constexpr size_t OFF_OUTWT = OFF_QKVWT + SZ_QKVWT;                // [768][768]
constexpr size_t SZ_OUTWT  = (size_t)HID * HID;
constexpr size_t OFF_PROJ  = OFF_OUTWT + SZ_OUTWT;                // [2][4096][1536] (q|k)
constexpr size_t SZ_PROJ   = (size_t)2 * M_TOT * NQK;
constexpr size_t OFF_VT    = OFF_PROJ + SZ_PROJ;                  // [2][4][12][64][1024]
constexpr size_t SZ_VT     = (size_t)2 * BATCH * HEADS * HDIM * SEQ;
constexpr size_t OFF_CTX   = OFF_VT + SZ_VT;                      // [2][4096][768]

// ---------------- operand loaders (ISA 7.12.2 layouts) ----------------
// A (16x32 bf16, row-major source, ld multiple of 8):
//   lanes 0-15 : row=lane,    K = {0..7, 16..23}
//   lanes16-31 : row=lane-16, K = {8..15, 24..31}
__device__ __forceinline__ v16bf load_a16(const bf16_t* tile, int ld) {
  const int lane = threadIdx.x & 31;
  const bf16_t* p = tile + (lane & 15) * ld + ((lane & 16) ? 8 : 0);
  v8bf lo = *(const v8bf*)(p);
  v8bf hi = *(const v8bf*)(p + 16);
  v16bf a;
#pragma unroll
  for (int i = 0; i < 8; i++) { a[i] = lo[i]; a[i + 8] = hi[i]; }
  return a;
}
// B (32x16 bf16) from an N-major source Bt[n][k] (ld multiple of 8):
//   lanes 0-15 : col=lane,    K = {0..7, 16..23} ; lanes 16-31 : col=lane-16, K = {8..15, 24..31}
__device__ __forceinline__ v16bf load_b16(const bf16_t* tile, int ld) {
  const int lane = threadIdx.x & 31;
  const bf16_t* p = tile + (lane & 15) * ld + ((lane & 16) ? 8 : 0);
  v8bf lo = *(const v8bf*)(p);
  v8bf hi = *(const v8bf*)(p + 16);
  v16bf b;
#pragma unroll
  for (int i = 0; i < 8; i++) { b[i] = lo[i]; b[i + 8] = hi[i]; }
  return b;
}

// ---------------- prep kernels ----------------
__global__ void __launch_bounds__(256) k_cvt(const float* __restrict__ src,
                                             bf16_t* __restrict__ dst, int n) {
  int i = blockIdx.x * 256 + threadIdx.x;
  if (i < n) dst[i] = (bf16_t)src[i];
}

// w[K][N] fp32 -> wt[N][K] bf16
__global__ void __launch_bounds__(256) k_transpose(const float* __restrict__ w,
                                                   bf16_t* __restrict__ wt, int K, int N) {
  int i = blockIdx.x * 256 + threadIdx.x;
  if (i < K * N) {
    int k = i / N, n = i - k * N;
    wt[(size_t)n * K + k] = (bf16_t)w[i];
  }
}

// ---------------- QKV GEMM: proj = Xbf @ qkv_w + b ----------------
// Block tile 128x128, 8 waves: wave = (wm in 0..3)*(wn in 0..1), each 2x4 WMMA tiles.
__global__ void __launch_bounds__(256) k_qkv_gemm(const bf16_t* __restrict__ xbf,
                                                  const bf16_t* __restrict__ wt,
                                                  const float* __restrict__ bias,
                                                  bf16_t* __restrict__ proj_qk,
                                                  bf16_t* __restrict__ vt) {
  const int mod = blockIdx.z;
  const int m0 = blockIdx.x * 128;
  const int n0 = blockIdx.y * 128;
  const int w = threadIdx.x >> 5;
  const int lane = threadIdx.x & 31;
  const int wm = (w & 3) * 32;   // 2 m-tiles
  const int wn = (w >> 2) * 64;  // 4 n-tiles

  const bf16_t* A  = xbf + (size_t)mod * M_TOT * HID + (size_t)(m0 + wm) * HID;
  const bf16_t* Bw = wt + (size_t)(n0 + wn) * HID;

  v8f acc[2][4];
#pragma unroll
  for (int i = 0; i < 2; i++)
#pragma unroll
    for (int j = 0; j < 4; j++) acc[i][j] = {};

  for (int k0 = 0; k0 < HID; k0 += 32) {
    v16bf a0 = load_a16(A + k0, HID);
    v16bf a1 = load_a16(A + 16 * HID + k0, HID);
#pragma unroll
    for (int j = 0; j < 4; j++) {
      v16bf b = load_b16(Bw + (size_t)j * 16 * HID + k0, HID);
      acc[0][j] = WMMA_BF16(a0, b, acc[0][j]);
      acc[1][j] = WMMA_BF16(a1, b, acc[1][j]);
    }
  }

  const int nlo = lane & 15;
  const int rbase = (lane & 16) ? 8 : 0;
#pragma unroll
  for (int i = 0; i < 2; i++) {
#pragma unroll
    for (int j = 0; j < 4; j++) {
      const int col = n0 + wn + j * 16 + nlo;
      const float bv = bias[col];
#pragma unroll
      for (int r = 0; r < 8; r++) {
        const int row = m0 + wm + i * 16 + rbase + r;  // [0,4096)
        const float v = acc[i][j][r] + bv;
        if (col < NQK) {  // q,k -> row-major
          proj_qk[((size_t)mod * M_TOT + row) * NQK + col] = (bf16_t)v;
        } else {          // v -> transposed Vt[mod][b][h][d][s]
          const int c2 = col - NQK;
          const int hh = c2 >> 6, dd = c2 & 63;
          const int bb = row >> 10, ss = row & 1023;
          vt[((((size_t)mod * BATCH + bb) * HEADS + hh) * HDIM + dd) * SEQ + ss] = (bf16_t)v;
        }
      }
    }
  }
}

// ---------------- attention: scores -> softmax -> blend -> PV ----------------
// Grid: (S/16 q-tiles, B*H). 256 threads = 8 waves. LDS: scores[2][16][1024] f32.
__global__ void __launch_bounds__(256) k_attn(const bf16_t* __restrict__ proj_qk,
                                              const bf16_t* __restrict__ vt,
                                              bf16_t* __restrict__ ctx) {
  extern __shared__ float smem[];
  float* sc   = smem;               // [2][16][1024]
  float* rmax = smem + 2 * 16 * 1024;  // [32]
  float* rsum = rmax + 32;             // [32]

  const int q0 = blockIdx.x * 16;
  const int b  = blockIdx.y / HEADS;
  const int h  = blockIdx.y % HEADS;
  const int w  = threadIdx.x >> 5;
  const int lane  = threadIdx.x & 31;
  const int nlo   = lane & 15;
  const int rbase = (lane & 16) ? 8 : 0;
  const size_t MODQK = (size_t)M_TOT * NQK;

  // ---- Phase 1: S = (Q Kt) * 1/sqrt(64), both modalities ----
  v16bf aq[2][2];
#pragma unroll
  for (int mod = 0; mod < 2; mod++) {
    const bf16_t* qb = proj_qk + mod * MODQK + (size_t)(b * SEQ + q0) * NQK + h * HDIM;
    aq[mod][0] = load_a16(qb, NQK);
    aq[mod][1] = load_a16(qb + 32, NQK);
  }
  for (int kt = w; kt < 64; kt += 8) {
#pragma unroll
    for (int mod = 0; mod < 2; mod++) {
      const bf16_t* kb =
          proj_qk + mod * MODQK + (size_t)(b * SEQ + kt * 16) * NQK + HID + h * HDIM;
      v16bf b0 = load_b16(kb, NQK);
      v16bf b1 = load_b16(kb + 32, NQK);
      v8f s = {};
      s = WMMA_BF16(aq[mod][0], b0, s);
      s = WMMA_BF16(aq[mod][1], b1, s);
      float* dst = sc + (mod * 16 + rbase) * 1024 + kt * 16 + nlo;
#pragma unroll
      for (int r = 0; r < 8; r++) dst[r * 1024] = s[r] * 0.125f;
    }
  }
  __syncthreads();

  // ---- Phase 2: row max / sumexp (32 rows, 4 per wave, wave32 shfl reduce) ----
  for (int g = w * 4; g < w * 4 + 4; g++) {
    const float* row = sc + g * 1024;
    float mx = -3.4e38f;
    for (int k = lane; k < 1024; k += 32) mx = fmaxf(mx, row[k]);
#pragma unroll
    for (int off = 16; off > 0; off >>= 1) mx = fmaxf(mx, __shfl_xor(mx, off, 32));
    float sum = 0.f;
    for (int k = lane; k < 1024; k += 32) sum += __expf(row[k] - mx);
#pragma unroll
    for (int off = 16; off > 0; off >>= 1) sum += __shfl_xor(sum, off, 32);
    if (lane == 0) { rmax[g] = mx; rsum[g] = sum; }
  }
  __syncthreads();

  // ---- Phase 3: normalize + cross-modal blend, in place (owner-exclusive) ----
  for (int idx = threadIdx.x; idx < 16 * 1024; idx += 256) {
    const int m = idx >> 10, k = idx & 1023;
    const float pi = __expf(sc[m * 1024 + k] - rmax[m]) / rsum[m];
    const float pd = __expf(sc[(16 + m) * 1024 + k] - rmax[16 + m]) / rsum[16 + m];
    sc[m * 1024 + k]        = 0.7f * pi + 0.3f * pd;  // shared_i
    sc[(16 + m) * 1024 + k] = 0.7f * pd + 0.3f * pi;  // shared_d
  }
  __syncthreads();

  // ---- Phase 4: ctx = P @ V (8 waves = 2 mods x 4 d-tiles) ----
  const int mod = w & 1;
  const int d0  = (w >> 1) * 16;
  const float* P = sc + mod * 16 * 1024;
  const bf16_t* vbase =
      vt + ((((size_t)mod * BATCH + b) * HEADS + h) * HDIM + d0) * SEQ;
  v8f acc = {};
  for (int ks = 0; ks < 1024; ks += 32) {
    const float* pp = P + (lane & 15) * 1024 + ks + ((lane & 16) ? 8 : 0);
    v16bf a;
#pragma unroll
    for (int i = 0; i < 8; i++) { a[i] = (bf16_t)pp[i]; a[i + 8] = (bf16_t)pp[16 + i]; }
    v16bf bv = load_b16(vbase + ks, SEQ);
    acc = WMMA_BF16(a, bv, acc);
  }
  bf16_t* cdst =
      ctx + ((size_t)mod * M_TOT + b * SEQ + q0 + rbase) * HID + h * HDIM + d0 + nlo;
#pragma unroll
  for (int r = 0; r < 8; r++) cdst[(size_t)r * HID] = (bf16_t)acc[r];
}

// ---------------- output GEMM: out = ctx @ out_w + out_b (fp32 store) ----------------
__global__ void __launch_bounds__(256) k_out_gemm(const bf16_t* __restrict__ ctx,
                                                  const bf16_t* __restrict__ wt,
                                                  const float* __restrict__ bias,
                                                  float* __restrict__ out) {
  const int mod = blockIdx.z;
  const int m0 = blockIdx.x * 128;
  const int n0 = blockIdx.y * 128;
  const int w = threadIdx.x >> 5;
  const int lane = threadIdx.x & 31;
  const int wm = (w & 3) * 32;
  const int wn = (w >> 2) * 64;

  const bf16_t* A  = ctx + (size_t)mod * M_TOT * HID + (size_t)(m0 + wm) * HID;
  const bf16_t* Bw = wt + (size_t)(n0 + wn) * HID;

  v8f acc[2][4];
#pragma unroll
  for (int i = 0; i < 2; i++)
#pragma unroll
    for (int j = 0; j < 4; j++) acc[i][j] = {};

  for (int k0 = 0; k0 < HID; k0 += 32) {
    v16bf a0 = load_a16(A + k0, HID);
    v16bf a1 = load_a16(A + 16 * HID + k0, HID);
#pragma unroll
    for (int j = 0; j < 4; j++) {
      v16bf b = load_b16(Bw + (size_t)j * 16 * HID + k0, HID);
      acc[0][j] = WMMA_BF16(a0, b, acc[0][j]);
      acc[1][j] = WMMA_BF16(a1, b, acc[1][j]);
    }
  }

  const int nlo = lane & 15;
  const int rbase = (lane & 16) ? 8 : 0;
  float* obase = out + (size_t)mod * M_TOT * HID;
#pragma unroll
  for (int i = 0; i < 2; i++) {
#pragma unroll
    for (int j = 0; j < 4; j++) {
      const int col = n0 + wn + j * 16 + nlo;
      const float bv = bias[col];
#pragma unroll
      for (int r = 0; r < 8; r++) {
        const int row = m0 + wm + i * 16 + rbase + r;
        obase[(size_t)row * HID + col] = acc[i][j][r] + bv;
      }
    }
  }
}

extern "C" void kernel_launch(void* const* d_in, const int* in_sizes, int n_in,
                              void* d_out, int out_size, void* d_ws, size_t ws_size,
                              hipStream_t stream) {
  const float* img   = (const float*)d_in[0];
  const float* dpt   = (const float*)d_in[1];
  const float* qkv_w = (const float*)d_in[2];
  const float* qkv_b = (const float*)d_in[3];
  const float* out_w = (const float*)d_in[4];
  const float* out_b = (const float*)d_in[5];

  bf16_t* ws     = (bf16_t*)d_ws;
  bf16_t* xbf    = ws + OFF_XBF;
  bf16_t* qkvwt  = ws + OFF_QKVWT;
  bf16_t* outwt  = ws + OFF_OUTWT;
  bf16_t* projqk = ws + OFF_PROJ;
  bf16_t* vt     = ws + OFF_VT;
  bf16_t* ctx    = ws + OFF_CTX;

  const int nx = M_TOT * HID;  // 3,145,728 per modality
  k_cvt<<<(nx + 255) / 256, 256, 0, stream>>>(img, xbf, nx);
  k_cvt<<<(nx + 255) / 256, 256, 0, stream>>>(dpt, xbf + (size_t)nx, nx);
  k_transpose<<<(HID * NQKV + 255) / 256, 256, 0, stream>>>(qkv_w, qkvwt, HID, NQKV);
  k_transpose<<<(HID * HID + 255) / 256, 256, 0, stream>>>(out_w, outwt, HID, HID);

  dim3 g1(M_TOT / 128, NQKV / 128, 2);  // 32 x 18 x 2
  k_qkv_gemm<<<g1, 256, 0, stream>>>(xbf, qkvwt, qkv_b, projqk, vt);

  dim3 g2(SEQ / 16, BATCH * HEADS, 1);  // 64 x 48
  const size_t shbytes = (size_t)(2 * 16 * 1024) * sizeof(float) + 64 * sizeof(float);
  k_attn<<<g2, 256, shbytes, stream>>>(projqk, vt, ctx);

  dim3 g3(M_TOT / 128, HID / 128, 2);   // 32 x 6 x 2
  k_out_gemm<<<g3, 256, 0, stream>>>(ctx, outwt, out_b, (float*)d_out);
}
